// SingleStreamODERNNPrototype_48713519071426
// MI455X (gfx1250) — compile-verified, matrix operand order checked by reading, not run on hardware
//
#include <hip/hip_runtime.h>
#include <math.h>

// ---------------------------------------------------------------------------
// ODE-RNN on MI455X (gfx1250), wave32 + WMMA bf16.
//
//  * Weights converted once to Wt (NxK bf16) -> B fragments are contiguous
//    16B loads; ~4.4MB, L2-resident (192MB L2).
//  * Activations that feed WMMA are kept in bf16 scratch, converted ONCE at
//    the producing epilogue (v_cvt_pk_bf16_f32 x4 per 8 outputs) instead of
//    at every A-fragment build -> inner GEMM loop is loads + wmma only.
//  * Fused encoder: 16-row strip per workgroup, layer-1 staged in LDS (bf16),
//    layer-2 WMMA reads LDS.
//  * Persistent scan kernel (32 blocks x 256 thr): 256 timesteps in one
//    launch, grid-wide sense barriers between dependent GEMM phases (cheaper
//    than ~1500 kernel launches; the scan is latency-bound, not BW-bound:
//    23.3TB/s HBM needs only ~17us for the 403MB of outputs).
// ---------------------------------------------------------------------------

#define BB   256
#define TT   256
#define FF   128
#define EE   256
#define HH   512
#define DECH 256
#define PP   128

typedef __attribute__((ext_vector_type(16))) __bf16 v16bf;
typedef __attribute__((ext_vector_type(8)))  __bf16 v8bf;
typedef __attribute__((ext_vector_type(8)))  float  v8f;

// ---- fragment loader -------------------------------------------------------
// 16-bit A/B layout (ISA 7.12.2): lane r=lane&15, half=lane>>4,
// frag elems e: K(e) = (e<8 ? e : e+8) + half*8 -> two contiguous 8-elem runs.
__device__ __forceinline__ void load_frag16(const __bf16* row, int k0, v16bf& f) {
  v8bf f0 = *(const v8bf*)(row + k0);
  v8bf f1 = *(const v8bf*)(row + k0 + 16);
#pragma unroll
  for (int e = 0; e < 8; ++e) { f[e] = f0[e]; f[e+8] = f1[e]; }
}

// C(16x16) = A(bf16, row stride lda) x Wt(N x K bf16, row-major), tile (mt,nt)
__device__ __forceinline__ v8f gemm_bb(const __bf16* A, long lda,
                                       const __bf16* Wt, int K,
                                       int mt, int nt) {
  const int lane = threadIdx.x & 31;
  const int l15  = lane & 15;
  const int half = lane >> 4;
  const __bf16* arow = A  + (long)(mt*16 + l15) * lda;
  const __bf16* brow = Wt + (long)(nt*16 + l15) * K;
  v8f c = {};
  for (int kc = 0; kc < K; kc += 32) {
    const int k0 = kc + (half << 3);
    v16bf a, b;
    load_frag16(arow, k0, a);
    load_frag16(brow, k0, b);
    c = __builtin_amdgcn_wmma_f32_16x16x32_bf16(false, a, false, b,
                                                (short)0, c, false, false);
  }
  return c;
}

__device__ __forceinline__ void cvt8(const float4 x0, const float4 x1, int off, v16bf& a) {
  a[off+0]=(__bf16)x0.x; a[off+1]=(__bf16)x0.y; a[off+2]=(__bf16)x0.z; a[off+3]=(__bf16)x0.w;
  a[off+4]=(__bf16)x1.x; a[off+5]=(__bf16)x1.y; a[off+6]=(__bf16)x1.z; a[off+7]=(__bf16)x1.w;
}

// ---- weight transpose+convert: Wt[n*K+k] = bf16(W[k*N+n]) ------------------
__global__ void wt_kernel(const float* __restrict__ W, __bf16* __restrict__ Wt,
                          int K, int N) {
  int i = blockIdx.x * blockDim.x + threadIdx.x;
  if (i < K * N) {
    int n = i / K, k = i - n * K;
    Wt[i] = (__bf16)W[(long)k * N + n];
  }
}

// ---- fused encoder ---------------------------------------------------------
__device__ __forceinline__ void enc_fill8(const float* vrow,
                                          const unsigned char* frow,
                                          int k0, int off, v16bf& a) {
  if (k0 < FF) {
    cvt8(*(const float4*)(vrow+k0), *(const float4*)(vrow+k0+4), off, a);
  } else {
    const unsigned char* p = frow + (k0 - FF);
#pragma unroll
    for (int e = 0; e < 8; ++e) a[off+e] = (__bf16)(p[e] ? 1.0f : 0.0f);
  }
}

__global__ void __launch_bounds__(256) enc_fused_kernel(
    const float* __restrict__ values, const unsigned char* __restrict__ fvm,
    const unsigned char* __restrict__ mask,
    const __bf16* __restrict__ w1t, const float* __restrict__ b1,
    const __bf16* __restrict__ w2t, const float* __restrict__ b2,
    float* __restrict__ emb, __bf16* __restrict__ embb) {
  __shared__ __bf16 H1b[16 * EE];               // 8KB bf16 strip, layer1 out
  const int strip = blockIdx.x;                 // rows strip*16 .. +15 of B*T
  const int lane  = threadIdx.x & 31;
  const int wave  = threadIdx.x >> 5;
  const int l15   = lane & 15;
  const int half  = lane >> 4;
  const long mrow = (long)strip * 16 + l15;
  const float*         vrow = values + mrow * FF;
  const unsigned char* frow = fvm    + mrow * FF;

  // layer 1: 8 waves x 2 N-tiles -> LDS (bf16). A built from f32/bool source.
#pragma unroll
  for (int j = 0; j < 2; ++j) {
    const int nt = wave * 2 + j;
    const __bf16* brow = w1t + (long)(nt*16 + l15) * (2*FF);
    v8f c = {};
    for (int kc = 0; kc < 2*FF; kc += 32) {
      const int k0 = kc + (half << 3);
      v16bf a, b;
      enc_fill8(vrow, frow, k0,      0, a);
      enc_fill8(vrow, frow, k0 + 16, 8, a);
      load_frag16(brow, k0, b);
      c = __builtin_amdgcn_wmma_f32_16x16x32_bf16(false, a, false, b,
                                                  (short)0, c, false, false);
    }
    const int n = nt*16 + l15, mb = half*8;
#pragma unroll
    for (int r = 0; r < 8; ++r) H1b[(mb + r) * EE + n] = (__bf16)tanhf(c[r] + b1[n]);
  }
  __syncthreads();

  // layer 2: pure bf16 WMMA from LDS; write fp32 output + bf16 shadow
#pragma unroll
  for (int j = 0; j < 2; ++j) {
    const int nt = wave * 2 + j;
    v8f c = gemm_bb(H1b, EE, w2t, EE, 0, nt);
    const int n = nt*16 + l15, mb = half*8;
#pragma unroll
    for (int r = 0; r < 8; ++r) {
      const long m  = (long)strip * 16 + mb + r;
      const float mf = mask[m] ? 1.0f : 0.0f;
      const float v  = (c[r] + b2[n]) * mf;
      emb[m * EE + n]  = v;
      embb[m * EE + n] = (__bf16)v;
    }
  }
}

// ---- persistent scan kernel ------------------------------------------------
struct ScanArgs {
  const float* dts; const unsigned char* mask;
  const __bf16* embb;                            // bf16 shadow of emb (B,T,E)
  const __bf16 *dyn_w1t, *dyn_w2t, *gru_wxt, *gru_wht,
               *dec_w1t, *dec_w2t, *proj_w1t, *proj_w2t;
  const float *dyn_b1, *dyn_b2, *gru_bx, *gru_bh,
              *dec_b1, *dec_b2, *proj_b1, *proj_b2;
  float  *h, *EV, *GX, *GH;                      // fp32 (elementwise consumers)
  __bf16 *hb, *EVb, *G1b, *D1b, *P1b;            // bf16 (GEMM A inputs)
  float *out_ev, *out_upd, *out_rec, *out_prj, *out_hf;
  unsigned *bar_cnt, *bar_gen;
  int nblocks;
};

__device__ __forceinline__ void grid_barrier(unsigned* cnt, unsigned* gen, int nblocks) {
  __syncthreads();
  if (threadIdx.x == 0) {
    __threadfence();
    const unsigned g = *(volatile unsigned*)gen;
    const unsigned arrived = atomicAdd(cnt, 1u);
    if (arrived == (unsigned)nblocks - 1u) {
      *(volatile unsigned*)cnt = 0u;
      __threadfence();
      atomicAdd(gen, 1u);
    } else {
      while (*(volatile unsigned*)gen == g) __builtin_amdgcn_s_sleep(1);
    }
    __threadfence();
  }
  __syncthreads();
}

__device__ __forceinline__ float sigm(float x) { return 1.0f / (1.0f + expf(-x)); }

__global__ void __launch_bounds__(256, 1) scan_kernel(ScanArgs A) {
  const int lane   = threadIdx.x & 31;
  const int l15    = lane & 15;
  const int half   = lane >> 4;
  const int gtid   = blockIdx.x * blockDim.x + threadIdx.x;
  const int nThr   = gridDim.x * blockDim.x;
  const int waveId = gtid >> 5;
  const int nWaves = nThr >> 5;

  for (int i = gtid; i < BB * HH; i += nThr) {          // h0 = 0 (both copies)
    A.h[i] = 0.0f; A.hb[i] = (__bf16)0.0f;
  }
  grid_barrier(A.bar_cnt, A.bar_gen, A.nblocks);

  for (int t = 0; t < TT; ++t) {
    // ---- A: G1b = bf16(tanh(h @ dyn_w1 + b1))   [256x512x512]
    for (int tile = waveId; tile < 16 * (HH/16); tile += nWaves) {
      const int mt = tile & 15, nt = tile >> 4;
      v8f c = gemm_bb(A.hb, HH, A.dyn_w1t, HH, mt, nt);
      const int n = nt*16 + l15, mb = mt*16 + half*8;
#pragma unroll
      for (int r = 0; r < 8; ++r)
        A.G1b[(mb+r)*HH + n] = (__bf16)tanhf(c[r] + A.dyn_b1[n]);
    }
    grid_barrier(A.bar_cnt, A.bar_gen, A.nblocks);

    // ---- B: ev = h + dt * tanh(G1 @ dyn_w2 + b2); EV fp32+bf16, masked out
    for (int tile = waveId; tile < 16 * (HH/16); tile += nWaves) {
      const int mt = tile & 15, nt = tile >> 4;
      v8f c = gemm_bb(A.G1b, HH, A.dyn_w2t, HH, mt, nt);
      const int n = nt*16 + l15, mb = mt*16 + half*8;
#pragma unroll
      for (int r = 0; r < 8; ++r) {
        const int m = mb + r;
        const float dyn = tanhf(c[r] + A.dyn_b2[n]);
        const float ev  = A.h[m*HH + n] + A.dts[(long)m*TT + t] * dyn;
        A.EV[m*HH + n]  = ev;
        A.EVb[m*HH + n] = (__bf16)ev;
        const float mf = A.mask[(long)m*TT + t] ? 1.0f : 0.0f;
        A.out_ev[((long)m*TT + t) * HH + n] = ev * mf;
      }
    }
    grid_barrier(A.bar_cnt, A.bar_gen, A.nblocks);

    // ---- C: GX = emb_t @ gru_wx + bx ; GH = ev @ gru_wh + bh  [N=1536]
    const int gxTiles = 16 * (3*HH/16);  // 1536
    for (int tile = waveId; tile < 2*gxTiles; tile += nWaves) {
      if (tile < gxTiles) {
        const int mt = tile & 15, nt = tile >> 4;
        v8f c = gemm_bb(A.embb + (long)t*EE, (long)TT*EE, A.gru_wxt, EE, mt, nt);
        const int n = nt*16 + l15, mb = mt*16 + half*8;
#pragma unroll
        for (int r = 0; r < 8; ++r) A.GX[(mb+r)*(3*HH) + n] = c[r] + A.gru_bx[n];
      } else {
        const int tl = tile - gxTiles;
        const int mt = tl & 15, nt = tl >> 4;
        v8f c = gemm_bb(A.EVb, HH, A.gru_wht, HH, mt, nt);
        const int n = nt*16 + l15, mb = mt*16 + half*8;
#pragma unroll
        for (int r = 0; r < 8; ++r) A.GH[(mb+r)*(3*HH) + n] = c[r] + A.gru_bh[n];
      }
    }
    grid_barrier(A.bar_cnt, A.bar_gen, A.nblocks);

    // ---- D: gates -> upd (= new h, fp32 + bf16), masked out_upd
    for (int i = gtid; i < BB * HH; i += nThr) {
      const int m = i >> 9, n = i & (HH - 1);
      const float xr = A.GX[m*3*HH + n],        hr = A.GH[m*3*HH + n];
      const float xz = A.GX[m*3*HH + HH + n],   hz = A.GH[m*3*HH + HH + n];
      const float xn = A.GX[m*3*HH + 2*HH + n], hn = A.GH[m*3*HH + 2*HH + n];
      const float rg = sigm(xr + hr);
      const float zg = sigm(xz + hz);
      const float ng = tanhf(xn + rg * hn);
      const float ev = A.EV[i];
      const float gr = (1.0f - zg) * ng + zg * ev;
      const bool  mk = A.mask[(long)m*TT + t] != 0;
      const float upd = mk ? gr : ev;
      A.h[i]  = upd;
      A.hb[i] = (__bf16)upd;
      A.out_upd[((long)m*TT + t) * HH + n] = mk ? upd : 0.0f;
    }
    grid_barrier(A.bar_cnt, A.bar_gen, A.nblocks);

    // ---- E: D1b = tanh(h @ dec_w1 + b) ; P1b = tanh(h @ proj_w1 + b)
    const int d1Tiles = 16 * (DECH/16);  // 256
    const int p1Tiles = 16 * (HH/16);    // 512
    for (int tile = waveId; tile < d1Tiles + p1Tiles; tile += nWaves) {
      if (tile < d1Tiles) {
        const int mt = tile & 15, nt = tile >> 4;
        v8f c = gemm_bb(A.hb, HH, A.dec_w1t, HH, mt, nt);
        const int n = nt*16 + l15, mb = mt*16 + half*8;
#pragma unroll
        for (int r = 0; r < 8; ++r)
          A.D1b[(mb+r)*DECH + n] = (__bf16)tanhf(c[r] + A.dec_b1[n]);
      } else {
        const int tl = tile - d1Tiles;
        const int mt = tl & 15, nt = tl >> 4;
        v8f c = gemm_bb(A.hb, HH, A.proj_w1t, HH, mt, nt);
        const int n = nt*16 + l15, mb = mt*16 + half*8;
#pragma unroll
        for (int r = 0; r < 8; ++r)
          A.P1b[(mb+r)*HH + n] = (__bf16)tanhf(c[r] + A.proj_b1[n]);
      }
    }
    grid_barrier(A.bar_cnt, A.bar_gen, A.nblocks);

    // ---- F: rec = (D1 @ dec_w2 + b)*mf ; prj = (P1 @ proj_w2 + b)*mf
    const int recTiles = 16 * (FF/16);   // 128
    const int prjTiles = 16 * (PP/16);   // 128
    for (int tile = waveId; tile < recTiles + prjTiles; tile += nWaves) {
      if (tile < recTiles) {
        const int mt = tile & 15, nt = tile >> 4;
        v8f c = gemm_bb(A.D1b, DECH, A.dec_w2t, DECH, mt, nt);
        const int n = nt*16 + l15, mb = mt*16 + half*8;
#pragma unroll
        for (int r = 0; r < 8; ++r) {
          const int m = mb + r;
          const float mf = A.mask[(long)m*TT + t] ? 1.0f : 0.0f;
          A.out_rec[((long)m*TT + t) * FF + n] = (c[r] + A.dec_b2[n]) * mf;
        }
      } else {
        const int tl = tile - recTiles;
        const int mt = tl & 15, nt = tl >> 4;
        v8f c = gemm_bb(A.P1b, HH, A.proj_w2t, HH, mt, nt);
        const int n = nt*16 + l15, mb = mt*16 + half*8;
#pragma unroll
        for (int r = 0; r < 8; ++r) {
          const int m = mb + r;
          const float mf = A.mask[(long)m*TT + t] ? 1.0f : 0.0f;
          A.out_prj[((long)m*TT + t) * PP + n] = (c[r] + A.proj_b2[n]) * mf;
        }
      }
    }
    grid_barrier(A.bar_cnt, A.bar_gen, A.nblocks);
  }

  for (int i = gtid; i < BB * HH; i += nThr) A.out_hf[i] = A.h[i];
}

// ---------------------------------------------------------------------------
extern "C" void kernel_launch(void* const* d_in, const int* in_sizes, int n_in,
                              void* d_out, int out_size, void* d_ws, size_t ws_size,
                              hipStream_t stream) {
  (void)in_sizes; (void)n_in; (void)out_size; (void)ws_size;
  const float*         values = (const float*)d_in[0];
  const unsigned char* mask   = (const unsigned char*)d_in[1];   // bool
  const unsigned char* fvm    = (const unsigned char*)d_in[2];   // bool
  const float*         dts    = (const float*)d_in[3];
  const float* enc_w1=(const float*)d_in[4],  *enc_b1=(const float*)d_in[5];
  const float* enc_w2=(const float*)d_in[6],  *enc_b2=(const float*)d_in[7];
  const float* dyn_w1=(const float*)d_in[8],  *dyn_b1=(const float*)d_in[9];
  const float* dyn_w2=(const float*)d_in[10], *dyn_b2=(const float*)d_in[11];
  const float* gru_wx=(const float*)d_in[12], *gru_wh=(const float*)d_in[13];
  const float* gru_bx=(const float*)d_in[14], *gru_bh=(const float*)d_in[15];
  const float* dec_w1=(const float*)d_in[16], *dec_b1=(const float*)d_in[17];
  const float* dec_w2=(const float*)d_in[18], *dec_b2=(const float*)d_in[19];
  const float* proj_w1=(const float*)d_in[20],*proj_b1=(const float*)d_in[21];
  const float* proj_w2=(const float*)d_in[22],*proj_b2=(const float*)d_in[23];

  // output sections (fp32), in reference return order
  float* out     = (float*)d_out;
  float* out_emb = out;
  float* out_ev  = out_emb + (long)BB*TT*EE;
  float* out_upd = out_ev  + (long)BB*TT*HH;
  float* out_rec = out_upd + (long)BB*TT*HH;
  float* out_prj = out_rec + (long)BB*TT*FF;
  float* out_hf  = out_prj + (long)BB*TT*PP;

  // workspace carve-up
  char* ws = (char*)d_ws;
  size_t off = 0;
  auto carve = [&](size_t bytes) -> char* {
    char* p = ws + off; off += (bytes + 255) & ~(size_t)255; return p;
  };
  unsigned* bar    = (unsigned*)carve(256);
  __bf16* enc_w1t  = (__bf16*)carve((size_t)2*FF*256*2);   // 256x256
  __bf16* enc_w2t  = (__bf16*)carve((size_t)EE*EE*2);
  __bf16* dyn_w1t  = (__bf16*)carve((size_t)HH*HH*2);
  __bf16* dyn_w2t  = (__bf16*)carve((size_t)HH*HH*2);
  __bf16* gru_wxt  = (__bf16*)carve((size_t)3*HH*EE*2);
  __bf16* gru_wht  = (__bf16*)carve((size_t)3*HH*HH*2);
  __bf16* dec_w1t  = (__bf16*)carve((size_t)DECH*HH*2);
  __bf16* dec_w2t  = (__bf16*)carve((size_t)FF*DECH*2);
  __bf16* proj_w1t = (__bf16*)carve((size_t)HH*HH*2);
  __bf16* proj_w2t = (__bf16*)carve((size_t)PP*HH*2);
  float*  h   = (float*) carve((size_t)BB*HH*4);
  float*  EV  = (float*) carve((size_t)BB*HH*4);
  float*  GX  = (float*) carve((size_t)BB*3*HH*4);
  float*  GH  = (float*) carve((size_t)BB*3*HH*4);
  __bf16* hb  = (__bf16*)carve((size_t)BB*HH*2);
  __bf16* EVb = (__bf16*)carve((size_t)BB*HH*2);
  __bf16* G1b = (__bf16*)carve((size_t)BB*HH*2);
  __bf16* D1b = (__bf16*)carve((size_t)BB*DECH*2);
  __bf16* P1b = (__bf16*)carve((size_t)BB*HH*2);
  __bf16* embb= (__bf16*)carve((size_t)BB*TT*EE*2);        // 32MB bf16 shadow

  hipMemsetAsync(bar, 0, 256, stream);

  // one-time weight transpose+bf16 convert (L2-resident afterwards)
  auto wt = [&](const float* W, __bf16* Wt, int K, int N) {
    int tot = K * N;
    wt_kernel<<<(tot + 255) / 256, 256, 0, stream>>>(W, Wt, K, N);
  };
  wt(enc_w1, enc_w1t, 2*FF, 256);
  wt(enc_w2, enc_w2t, EE, EE);
  wt(dyn_w1, dyn_w1t, HH, HH);
  wt(dyn_w2, dyn_w2t, HH, HH);
  wt(gru_wx, gru_wxt, EE, 3*HH);
  wt(gru_wh, gru_wht, HH, 3*HH);
  wt(dec_w1, dec_w1t, HH, DECH);
  wt(dec_w2, dec_w2t, DECH, FF);
  wt(proj_w1, proj_w1t, HH, HH);
  wt(proj_w2, proj_w2t, HH, PP);

  // encoder: one 16-row strip per workgroup over B*T rows
  enc_fused_kernel<<<(BB*TT)/16, 256, 0, stream>>>(
      values, fvm, mask, enc_w1t, enc_b1, enc_w2t, enc_b2, out_emb, embb);

  // persistent scan
  ScanArgs A;
  A.dts = dts; A.mask = mask; A.embb = embb;
  A.dyn_w1t = dyn_w1t; A.dyn_w2t = dyn_w2t;
  A.gru_wxt = gru_wxt; A.gru_wht = gru_wht;
  A.dec_w1t = dec_w1t; A.dec_w2t = dec_w2t;
  A.proj_w1t = proj_w1t; A.proj_w2t = proj_w2t;
  A.dyn_b1 = dyn_b1; A.dyn_b2 = dyn_b2;
  A.gru_bx = gru_bx; A.gru_bh = gru_bh;
  A.dec_b1 = dec_b1; A.dec_b2 = dec_b2;
  A.proj_b1 = proj_b1; A.proj_b2 = proj_b2;
  A.h = h; A.EV = EV; A.GX = GX; A.GH = GH;
  A.hb = hb; A.EVb = EVb; A.G1b = G1b; A.D1b = D1b; A.P1b = P1b;
  A.out_ev = out_ev; A.out_upd = out_upd; A.out_rec = out_rec;
  A.out_prj = out_prj; A.out_hf = out_hf;
  A.bar_cnt = bar; A.bar_gen = bar + 1;
  A.nblocks = 32;
  scan_kernel<<<32, 256, 0, stream>>>(A);
}
